// QuantizedD4Linear_15814069584511
// MI455X (gfx1250) — compile-verified, matrix operand order checked by reading, not run on hardware
//
#include <hip/hip_runtime.h>
#include <hip/hip_bf16.h>

typedef __attribute__((ext_vector_type(16))) _Float16 v16h;
typedef __attribute__((ext_vector_type(8)))  float    v8f;

#define T_TOK   8192
#define K_DIM   4096
#define M_DIM   4096
#define TILE_T  128
#define TILE_M  128
#define KSTEP   64
#define NSTEP   (K_DIM / KSTEP)          // 64
#define LDSROW  72                        // 64 halves + 8 pad -> 144B stride, conflict-free
#define BUFELT  (TILE_M * LDSROW)         // halves per buffer
#define BUFBYTES (BUFELT * 2)             // 18432 B
#define NBUF    3                         // 55296 B total LDS

// ---------------- Kernel 1: x = fwht(input * SU) / (64*1024) -> fp16 ----------------
__global__ __launch_bounds__(256) void k_fwht_in(const float* __restrict__ in,
                                                 const float* __restrict__ SU,
                                                 _Float16* __restrict__ Xh) {
    __shared__ float buf[K_DIM];
    const int row = blockIdx.x;
    const float* src = in + (size_t)row * K_DIM;
    for (int i = threadIdx.x; i < K_DIM; i += 256)
        buf[i] = src[i] * SU[i];
    __syncthreads();
    for (int h = 1; h < K_DIM; h <<= 1) {
        for (int p = threadIdx.x; p < (K_DIM / 2); p += 256) {
            int i = ((p & ~(h - 1)) << 1) | (p & (h - 1));
            int j = i + h;
            float a = buf[i], b = buf[j];
            buf[i] = a + b;
            buf[j] = a - b;
        }
        __syncthreads();
    }
    const float s = 1.0f / (64.0f * 1024.0f);   // 1/sqrt(4096) * 1/1024
    _Float16* dst = Xh + (size_t)row * K_DIM;
    for (int i = threadIdx.x; i < K_DIM; i += 256)
        dst[i] = (_Float16)(buf[i] * s);
}

// ---------------- Kernel 2: W16[i,0:4] = D4_CB[Qidxs[i]] ----------------
__global__ __launch_bounds__(256) void k_decompress(const int* __restrict__ Qidxs,
                                                    const _Float16* __restrict__ CB,
                                                    _Float16* __restrict__ W16,
                                                    int total) {
    int idx = blockIdx.x * 256 + threadIdx.x;
    if (idx >= total) return;
    int q = Qidxs[idx] & 255;
    uint2 v = *(const uint2*)(CB + q * 4);          // 4 fp16 lattice points
    *(uint2*)(W16 + (size_t)idx * 4) = v;
}

// ---------------- Kernel 3: Z16 = Xh @ W16^T (fp16 WMMA, async-LDS pipeline) --------
// Block: 128x128 tile, 8 waves. Per wave: 32 tokens x 64 m = 2 A-frags x 4 B-frags,
// two K sub-steps (k, k+32) per iteration => 16 WMMA between barriers.
__global__ __launch_bounds__(256) void k_gemm(const _Float16* __restrict__ Xh,
                                              const _Float16* __restrict__ W16,
                                              _Float16* __restrict__ Z16) {
    __shared__ _Float16 bsh[NBUF * BUFELT];

    const int tid    = threadIdx.x;
    const int wave   = tid >> 5;
    const int lane   = tid & 31;
    const int lhalf  = lane >> 4;     // 0: lanes 0-15, 1: lanes 16-31
    const int l16    = lane & 15;
    const int wave_t = wave & 3;      // 4 token strips of 32
    const int wave_m = wave >> 2;     // 2 m strips of 64

    const int t0 = blockIdx.x * TILE_T;
    const int m0 = blockIdx.y * TILE_M;

    v8f acc[2][4] = {};   // [A-frag][B-frag] 16x16 f32 tiles

    // ---- async B staging: 128 rows x 64 halves = 1024 b128 chunks, 4 per thread ----
    const int r0  = tid >> 3;            // rows 0..31; +32/+64/+96 for the other chunks
    const int cc0 = (tid & 7) * 8;       // half offset within the 64-wide K slab
    const _Float16* gbase = W16 + (size_t)(m0 + r0) * K_DIM + cc0;
    const unsigned ldsbase = (unsigned)(uintptr_t)(&bsh[r0 * LDSROW + cc0]);

    // issue four async b128 copies (16B each) for K-step s into buffer at byte off
    auto issue_tile = [&](int s, unsigned bufoff) {
        const unsigned  lo = ldsbase + bufoff;
        const _Float16* gp = gbase + (size_t)s * KSTEP;
#pragma unroll
        for (int j = 0; j < 4; ++j) {
            asm volatile("global_load_async_to_lds_b128 %0, %1, off"
                         :: "v"(lo + (unsigned)(j * 32 * LDSROW * 2)),
                            "v"(gp + (size_t)(j * 32) * K_DIM)
                         : "memory");
        }
    };

    // 16-bit A layout: lanes 0-15 hold K {0..7,16..23}, lanes 16-31 hold K {8..15,24..31}
    const _Float16* aBase0 = Xh + (size_t)(t0 + wave_t * 32 + l16) * K_DIM + lhalf * 8;
    const _Float16* aBase1 = aBase0 + (size_t)16 * K_DIM;

    auto load_a = [&](v16h& a0, v16h& a1, int koff) {
        const _Float16* p0 = aBase0 + koff;
        const _Float16* p1 = aBase1 + koff;
        ((uint4*)&a0)[0] = *(const uint4*)(p0);
        ((uint4*)&a0)[1] = *(const uint4*)(p0 + 16);
        ((uint4*)&a1)[0] = *(const uint4*)(p1);
        ((uint4*)&a1)[1] = *(const uint4*)(p1 + 16);
    };

    // LDS read offset within a buffer (halves)
    const int rdoff = (wave_m * 64 + l16) * LDSROW + lhalf * 8;

    // one K-step: consume cur (A at s*KSTEP), prefetch A((s+1)*KSTEP) into nxt.
    // comp_off: compute buffer offset in halves; iss_off: issue buffer offset in bytes.
    auto step = [&](int s, v16h& cur0, v16h& cur1, v16h& nxt0, v16h& nxt1,
                    unsigned& comp_off, unsigned& iss_off) {
        // FIFO: <=4 outstanding leaves only tile s+1's 4 loads -> tile s resident
        if (s + 1 < NSTEP) asm volatile("s_wait_asynccnt 0x4" ::: "memory");
        else               asm volatile("s_wait_asynccnt 0x0" ::: "memory");
        __syncthreads();   // tile s visible; all reads of tile s-1 done

        if (s + 2 < NSTEP) issue_tile(s + 2, iss_off);   // last read at s-1

        // in-iteration A pair for sub-step 1 (k+32), hidden behind sub-step 0 WMMAs
        v16h mid0, mid1;
        load_a(mid0, mid1, s * KSTEP + 32);
        // cross-iteration A prefetch (ping-pong registers: no rotation copies)
        const int sn = (s + 1 < NSTEP) ? s + 1 : s;
        load_a(nxt0, nxt1, sn * KSTEP);

        const _Float16* bb = &bsh[comp_off + rdoff];
        auto loadB = [&](v16h& b, int p) {   // p: 0..7 -> (n = p&3, ksub = p>>2)
            const _Float16* bp = bb + (p & 3) * 16 * LDSROW + (p >> 2) * 32;
            ((uint4*)&b)[0] = *(const uint4*)(bp);
            ((uint4*)&b)[1] = *(const uint4*)(bp + 16);
        };
        auto wmma2 = [&](int n, const v16h& a0, const v16h& a1, const v16h& b) {
            acc[0][n] = __builtin_amdgcn_wmma_f32_16x16x32_f16(
                false, a0, false, b, (short)0, acc[0][n], false, false);
            acc[1][n] = __builtin_amdgcn_wmma_f32_16x16x32_f16(
                false, a1, false, b, (short)0, acc[1][n], false, false);
        };

        // explicit B ping-pong: load pair i+1 issued before WMMAs of pair i
        v16h bA, bB;
        loadB(bA, 0);
        loadB(bB, 1); wmma2(0, cur0, cur1, bA);
        loadB(bA, 2); wmma2(1, cur0, cur1, bB);
        loadB(bB, 3); wmma2(2, cur0, cur1, bA);
        loadB(bA, 4); wmma2(3, cur0, cur1, bB);
        loadB(bB, 5); wmma2(0, mid0, mid1, bA);
        loadB(bA, 6); wmma2(1, mid0, mid1, bB);
        loadB(bB, 7); wmma2(2, mid0, mid1, bA);
                      wmma2(3, mid0, mid1, bB);

        // rotate buffers without integer division
        comp_off = (comp_off == 2u * BUFELT)   ? 0u : comp_off + BUFELT;
        iss_off  = (iss_off  == 2u * BUFBYTES) ? 0u : iss_off + BUFBYTES;
    };

    // ---- prologue: 2 async B tiles in flight, A(0) loaded ----
    issue_tile(0, 0u);
    issue_tile(1, BUFBYTES);
    unsigned comp_off = 0u;                 // halves: buffer 0
    unsigned iss_off  = 2u * BUFBYTES;      // bytes:  buffer 2 (target of s+2 at s=0)

    v16h afA0, afA1, afB0, afB1;
    load_a(afA0, afA1, 0);

    for (int s = 0; s < NSTEP; s += 2) {
        step(s,     afA0, afA1, afB0, afB1, comp_off, iss_off);
        step(s + 1, afB0, afB1, afA0, afA1, comp_off, iss_off);
    }

    // epilogue: C/D layout -> row M = r + 8*lhalf, col N = l16
#pragma unroll
    for (int t = 0; t < 2; ++t) {
#pragma unroll
        for (int n = 0; n < 4; ++n) {
#pragma unroll
            for (int r = 0; r < 8; ++r) {
                int trow = t0 + wave_t * 32 + t * 16 + r + lhalf * 8;
                int m    = m0 + wave_m * 64 + n * 16 + l16;
                Z16[(size_t)trow * M_DIM + m] = (_Float16)acc[t][n][r];
            }
        }
    }
}

// ---------------- Kernel 4: out = fwht(Z16 * Wscale*1024) * SV ----------------
__global__ __launch_bounds__(256) void k_fwht_out(const _Float16* __restrict__ Z16,
                                                  const float* __restrict__ SV,
                                                  const float* __restrict__ Wscale,
                                                  float* __restrict__ out) {
    __shared__ float buf[M_DIM];
    const int row = blockIdx.x;
    const _Float16* src = Z16 + (size_t)row * M_DIM;
    const float ws = Wscale[0] * 1024.0f;
    for (int i = threadIdx.x; i < M_DIM; i += 256)
        buf[i] = (float)src[i] * ws;
    __syncthreads();
    for (int h = 1; h < M_DIM; h <<= 1) {
        for (int p = threadIdx.x; p < (M_DIM / 2); p += 256) {
            int i = ((p & ~(h - 1)) << 1) | (p & (h - 1));
            int j = i + h;
            float a = buf[i], b = buf[j];
            buf[i] = a + b;
            buf[j] = a - b;
        }
        __syncthreads();
    }
    const float s = 1.0f / 64.0f;   // 1/sqrt(4096)
    float* dst = out + (size_t)row * M_DIM;
    for (int i = threadIdx.x; i < M_DIM; i += 256)
        dst[i] = buf[i] * s * SV[i];
}

extern "C" void kernel_launch(void* const* d_in, const int* in_sizes, int n_in,
                              void* d_out, int out_size, void* d_ws, size_t ws_size,
                              hipStream_t stream) {
    const float*    input  = (const float*)d_in[0];
    const int*      Qidxs  = (const int*)d_in[1];
    const float*    SU     = (const float*)d_in[2];
    const float*    SV     = (const float*)d_in[3];
    const float*    Wscale = (const float*)d_in[4];
    const _Float16* CB     = (const _Float16*)d_in[9];
    float*          out    = (float*)d_out;

    char* ws = (char*)d_ws;
    _Float16* Xh  = (_Float16*)(ws);                                   // 64 MB
    _Float16* W16 = (_Float16*)(ws + (size_t)64 * 1024 * 1024);        // 32 MB
    _Float16* Z16 = (_Float16*)(ws + (size_t)96 * 1024 * 1024);        // 64 MB

    k_fwht_in<<<T_TOK, 256, 0, stream>>>(input, SU, Xh);

    const int total = M_DIM * (K_DIM / 4);   // 4096 * 1024 codes
    k_decompress<<<(total + 255) / 256, 256, 0, stream>>>(Qidxs, CB, W16, total);

    dim3 grid(T_TOK / TILE_T, M_DIM / TILE_M);
    k_gemm<<<grid, 256, 0, stream>>>(Xh, W16, Z16);

    k_fwht_out<<<T_TOK, 256, 0, stream>>>(Z16, SV, Wscale, out);
}
